// InstAtt_26096221291299
// MI455X (gfx1250) — matrix operand order repaired and checked by reading, not compiled, not run on hardware
//
#include <hip/hip_runtime.h>
#include <hip/hip_bf16.h>

typedef __attribute__((ext_vector_type(16))) _Float16 v16h;
typedef __attribute__((ext_vector_type(8)))  _Float16 v8h;
typedef __attribute__((ext_vector_type(4)))  _Float16 v4h;
typedef __attribute__((ext_vector_type(8)))  float    v8f;
typedef __attribute__((ext_vector_type(4)))  float    v4f;

#define B_TOT 512
#define T_SEQ 64
#define D_HID 64
#define M_MEM 50
#define V_OUT 2048
#define KCTX  96    // 68 padded to 3 K-chunks of 32
#define KGATE 128   // [x(64) | h(64)]

// fast transcendental forms: v_exp_f32 + v_add + v_rcp_f32 (no precise-div expansion)
__device__ __forceinline__ float fast_sigmoid(float x) {
  return __builtin_amdgcn_rcpf(1.0f + __expf(-x));
}
__device__ __forceinline__ float fast_tanh(float x) {
  return 2.0f * __builtin_amdgcn_rcpf(1.0f + __expf(-2.0f * x)) - 1.0f;
}

// ---- A fragment (16x32 f16): lane<16 -> row=lane, K={0..7,16..23}; lane>=16 -> K={8..15,24..31}
// Two contiguous 8-half runs -> 2 x b128 loads.
__device__ __forceinline__ v16h load_a_frag(const _Float16* A, int lda, int k0, int lane) {
  int row = lane & 15;
  int kb  = k0 + ((lane & 16) ? 8 : 0);
  const _Float16* p = A + row * lda + kb;
  union { v16h v; struct { v8h lo, hi; } s; } u;
  u.s.lo = *(const v8h*)(p);
  u.s.hi = *(const v8h*)(p + 16);
  return u.v;
}
// ---- B fragment from pre-swizzled storage: frag (kc,nt) = 32 lanes x 16 halves contiguous.
__device__ __forceinline__ v16h load_b_packed(const _Float16* P, int NT, int kc, int nt, int lane) {
  return *(const v16h*)(P + ((((kc * NT) + nt) * 32 + lane) << 4));
}

// ============================ prep: memory tensor + packed f16 weights + ctx clear ============================
__global__ void prep_kernel(const float* __restrict__ prev, const float* __restrict__ fin,
                            const float* __restrict__ W_ih, const float* __restrict__ W_hh,
                            const float* __restrict__ b_ih, const float* __restrict__ b_hh,
                            const float* __restrict__ W_att1, const float* __restrict__ W_out,
                            float* __restrict__ memB, _Float16* __restrict__ WgP,
                            float* __restrict__ bg, _Float16* __restrict__ Wa1P,
                            _Float16* __restrict__ WoutP, _Float16* __restrict__ ctx) {
  int tid0 = blockIdx.x * blockDim.x + threadIdx.x;
  int stride = gridDim.x * blockDim.x;
  // memory = concat(where(mask,-1,final), prev) : (B, 50, 4)
  for (int e = tid0; e < B_TOT * 25; e += stride) {
    int b = e / 25, m = e % 25;
    const float* pc = prev + (b * 25 + m) * 4;
    const float* fc = fin + (b * 25 + m) * 4;
    float s = pc[0] + pc[1] + pc[2] + pc[3];
    bool mask = s > 0.0f;
    float* d0 = memB + (b * 50 + m) * 4;
    float* d1 = memB + (b * 50 + 25 + m) * 4;
#pragma unroll
    for (int f = 0; f < 4; ++f) { d0[f] = mask ? -1.0f : fc[f]; d1[f] = pc[f]; }
  }
  // gate weights, packed: KC=4, NT=16
  for (int e = tid0; e < 4 * 16 * 512; e += stride) {
    int j = e & 15, lane = (e >> 4) & 31, rest = e >> 9;
    int nt = rest & 15, kc = rest >> 4;
    int k = kc * 32 + ((lane & 16) ? 8 : 0) + (j < 8 ? j : j + 8);
    int n = nt * 16 + (lane & 15);
    float v = (k < 64) ? W_ih[n * 64 + k] : W_hh[n * 64 + (k - 64)];
    WgP[e] = (_Float16)v;
  }
  for (int e = tid0; e < 256; e += stride) bg[e] = b_ih[e] + b_hh[e];
  // attention weights, packed: KC=3, NT=2 (pad K 68->96)
  for (int e = tid0; e < 3 * 2 * 512; e += stride) {
    int j = e & 15, lane = (e >> 4) & 31, rest = e >> 9;
    int nt = rest & 1, kc = rest >> 1;
    int k = kc * 32 + ((lane & 16) ? 8 : 0) + (j < 8 ? j : j + 8);
    int n = nt * 16 + (lane & 15);
    Wa1P[e] = (_Float16)((k < 68) ? W_att1[n * 68 + k] : 0.0f);
  }
  // output weights, packed: KC=3, NT=128 (pad K 68->96)
  for (int e = tid0; e < 3 * 128 * 512; e += stride) {
    int j = e & 15, lane = (e >> 4) & 31, rest = e >> 9;
    int nt = rest & 127, kc = rest >> 7;
    int k = kc * 32 + ((lane & 16) ? 8 : 0) + (j < 8 ? j : j + 8);
    int n = nt * 16 + (lane & 15);
    WoutP[e] = (_Float16)((k < 68) ? W_out[n * 68 + k] : 0.0f);
  }
  // zero ctx once (recurrence rewrites k=0..67 every call; pad k=68..95 stays 0)
  unsigned long long* cz = (unsigned long long*)ctx;
  for (int e = tid0; e < (B_TOT * T_SEQ * KCTX) / 4; e += stride) cz[e] = 0ull;
}

// ============================ recurrence: 32 WGs x 16 batch rows, 64 steps ============================
__global__ __launch_bounds__(256) void recur_kernel(
    const int* __restrict__ inst, const float* __restrict__ embed,
    const _Float16* __restrict__ WgP, const float* __restrict__ bg,
    const _Float16* __restrict__ Wa1P, const float* __restrict__ b_att1,
    const float* __restrict__ W_att2, const float* __restrict__ b_att2,
    const float* __restrict__ memB, _Float16* __restrict__ ctx) {
  __shared__ _Float16 sA[16 * KGATE];       // [x | h] f16 per batch row
  __shared__ float    sC[16 * D_HID];       // cell state
  __shared__ float    sG[16 * 256];         // gate pre-activations
  __shared__ float    sMem[16 * M_MEM * 4];
  __shared__ float    sScore[16 * M_MEM];
  __shared__ float    sAtt[16 * 4];
  __shared__ int      sInst[16 * T_SEQ];    // per-WG instruction tile, staged once

  const int tid = threadIdx.x;
  const int lane = tid & 31;
  const int wave = tid >> 5;
  const int b0 = blockIdx.x * 16;
  const int nl = lane & 15;
  const int mh = (lane & 16) ? 8 : 0;

  // ---- weight fragments live in registers across all 64 steps ----
  v16h bgF[2][4];   // gates: this wave handles n-tiles (wave) and (wave+8)
#pragma unroll
  for (int q = 0; q < 2; ++q)
#pragma unroll
    for (int kc = 0; kc < 4; ++kc) bgF[q][kc] = load_b_packed(WgP, 16, kc, wave + q * 8, lane);
  v16h baF[2][3];   // attention: both n-tiles
#pragma unroll
  for (int q = 0; q < 2; ++q)
#pragma unroll
    for (int kc = 0; kc < 3; ++kc) baF[q][kc] = load_b_packed(Wa1P, 2, kc, q, lane);
  float bgv[2];
#pragma unroll
  for (int q = 0; q < 2; ++q) bgv[q] = bg[(wave + q * 8) * 16 + nl];
  const float w2a = W_att2[nl], w2b = W_att2[16 + nl];
  const float ba1a = b_att1[nl], ba1b = b_att1[16 + nl];
  const float ba2 = b_att2[0];

  for (int c = tid; c < 16 * M_MEM; c += 256)
    *(v4f*)&sMem[c * 4] = *(const v4f*)&memB[b0 * M_MEM * 4 + c * 4];
  for (int e = tid; e < 16 * T_SEQ; e += 256) {
    int r = e >> 6, tt = e & 63;
    sInst[e] = inst[(b0 + r) * 65 + tt];
  }
  for (int e = tid; e < 16 * D_HID; e += 256) sC[e] = 0.0f;
  for (int e = tid; e < 16 * KGATE; e += 256) sA[e] = (_Float16)0.0f;
  __syncthreads();

  for (int t = 0; t < T_SEQ; ++t) {
    // 1) embedding lookup -> sA[:, 0:64] (index from LDS; float4 -> half4)
    {
      int r = tid >> 4, dc = tid & 15;
      int idx = sInst[r * 64 + t];
      v4f x = *(const v4f*)&embed[idx * D_HID + dc * 4];
      v4h hx;
#pragma unroll
      for (int f = 0; f < 4; ++f) hx[f] = (_Float16)x[f];
      *(v4h*)&sA[r * KGATE + dc * 4] = hx;
    }
    __syncthreads();
    // 2) gates = [x|h] @ [W_ih;W_hh]^T : M=16, N=256, K=128 via WMMA
    {
      v16h aF[4];
#pragma unroll
      for (int kc = 0; kc < 4; ++kc) aF[kc] = load_a_frag(sA, KGATE, kc * 32, lane);
#pragma unroll
      for (int q = 0; q < 2; ++q) {
        v8f acc = {};
#pragma unroll
        for (int kc = 0; kc < 4; ++kc)
          acc = __builtin_amdgcn_wmma_f32_16x16x32_f16(false, aF[kc], false, bgF[q][kc], (short)0, acc, false, false);
        int n = (wave + q * 8) * 16 + nl;
#pragma unroll
        for (int i = 0; i < 8; ++i) sG[(i + mh) * 256 + n] = acc[i] + bgv[q];
      }
    }
    __syncthreads();
    // 3) LSTM pointwise; h -> sA[:, 64:128] (f16)
    for (int e = tid; e < 16 * D_HID; e += 256) {
      int r = e >> 6, d = e & 63;
      float ig = sG[r * 256 + d];
      float fg = sG[r * 256 + 64 + d];
      float gg = sG[r * 256 + 128 + d];
      float og = sG[r * 256 + 192 + d];
      float c = fast_sigmoid(fg) * sC[e] + fast_sigmoid(ig) * fast_tanh(gg);
      sC[e] = c;
      float h = fast_sigmoid(og) * fast_tanh(c);
      sA[r * KGATE + 64 + d] = (_Float16)h;
    }
    __syncthreads();
    // 4) attention scores: 800 rows (16b x 50m), K=68->96, N=32; fuse relu+W_att2+lane reduce
    for (int ti = wave; ti < 50; ti += 8) {
      int ra = ti * 16 + nl;
      int bi = ra / 50, m = ra % 50;
      v16h a0, a1, a2;
      { // k-chunks 0,1 are pure h: contiguous 8-half runs from sA
        const _Float16* ph = &sA[bi * KGATE + 64];
        union { v16h v; struct { v8h lo, hi; } s; } u0, u1;
        int kb0 = (lane & 16) ? 8 : 0;
        u0.s.lo = *(const v8h*)(ph + kb0);
        u0.s.hi = *(const v8h*)(ph + kb0 + 16);
        u1.s.lo = *(const v8h*)(ph + 32 + kb0);
        u1.s.hi = *(const v8h*)(ph + 32 + kb0 + 16);
        a0 = u0.v; a1 = u1.v;
      }
      { // k-chunk 2: K=64..95 -> 4 memory dims then zero pad (only lanes<16 carry data)
#pragma unroll
        for (int j = 0; j < 16; ++j) a2[j] = (_Float16)0.0f;
        if (lane < 16) {
          const float* pm = &sMem[bi * 200 + m * 4];
#pragma unroll
          for (int f = 0; f < 4; ++f) a2[f] = (_Float16)pm[f];
        }
      }
      v8f acc0 = {}, acc1 = {};
      acc0 = __builtin_amdgcn_wmma_f32_16x16x32_f16(false, a0, false, baF[0][0], (short)0, acc0, false, false);
      acc1 = __builtin_amdgcn_wmma_f32_16x16x32_f16(false, a0, false, baF[1][0], (short)0, acc1, false, false);
      acc0 = __builtin_amdgcn_wmma_f32_16x16x32_f16(false, a1, false, baF[0][1], (short)0, acc0, false, false);
      acc1 = __builtin_amdgcn_wmma_f32_16x16x32_f16(false, a1, false, baF[1][1], (short)0, acc1, false, false);
      acc0 = __builtin_amdgcn_wmma_f32_16x16x32_f16(false, a2, false, baF[0][2], (short)0, acc0, false, false);
      acc1 = __builtin_amdgcn_wmma_f32_16x16x32_f16(false, a2, false, baF[1][2], (short)0, acc1, false, false);
#pragma unroll
      for (int i = 0; i < 8; ++i) {
        float v = fmaxf(acc0[i] + ba1a, 0.0f) * w2a + fmaxf(acc1[i] + ba1b, 0.0f) * w2b;
#pragma unroll
        for (int off = 1; off < 16; off <<= 1) v += __shfl_xor(v, off, 32);
        if (nl == 0) sScore[ti * 16 + i + mh] = v + ba2;
      }
    }
    __syncthreads();
    // 5a) softmax over M=50 per batch row
    if (tid < 16) {
      int bi = tid;
      float mx = -3.4e38f;
      for (int m = 0; m < M_MEM; ++m) mx = fmaxf(mx, sScore[bi * M_MEM + m]);
      float s = 0.0f;
      for (int m = 0; m < M_MEM; ++m) {
        float e2 = __expf(sScore[bi * M_MEM + m] - mx);
        sScore[bi * M_MEM + m] = e2;
        s += e2;
      }
      float inv = __builtin_amdgcn_rcpf(s);
      for (int m = 0; m < M_MEM; ++m) sScore[bi * M_MEM + m] *= inv;
    }
    __syncthreads();
    // 5b) att = weight @ memory
    if (tid < 64) {
      int bi = tid >> 2, f = tid & 3;
      float a = 0.0f;
      for (int m = 0; m < M_MEM; ++m) a += sScore[bi * M_MEM + m] * sMem[bi * 200 + m * 4 + f];
      sAtt[tid] = a;
    }
    __syncthreads();
    // 6) stage ctx = [h | att] f16 (pad was zeroed once in prep); one b64 per thread + att rows.
    //    Reads only sA h-region/sAtt; next step's phase 1 writes only the x-region -> no barrier needed.
    {
      int r = tid >> 4, kk = (tid & 15) * 4;
      long long base = ((long long)(b0 + r) * T_SEQ + t) * KCTX;
      *(v4h*)&ctx[base + kk] = *(const v4h*)&sA[r * KGATE + 64 + kk];
      if (tid < 16) {
        v4h v;
#pragma unroll
        for (int f = 0; f < 4; ++f) v[f] = (_Float16)sAtt[tid * 4 + f];
        *(v4h*)&ctx[((long long)(b0 + tid) * T_SEQ + t) * KCTX + 64] = v;
      }
    }
  }
}

// ============================ logits + log_softmax: 2048 WGs x 16 rows ============================
__global__ __launch_bounds__(256) void logits_kernel(
    const _Float16* __restrict__ ctx, const _Float16* __restrict__ WoutP,
    const float* __restrict__ b_out, float* __restrict__ out) {
  __shared__ _Float16 sA[16 * KCTX];    // 3 KB
  __shared__ float    sLog[16 * V_OUT]; // 128 KB
  const int tid = threadIdx.x, lane = tid & 31, wave = tid >> 5;
  const int nl = lane & 15, mh = (lane & 16) ? 8 : 0;
  const long long r0 = (long long)blockIdx.x * 16;

  // warm GL2 for this wave's W_out fragment stream (global_prefetch_b8)
#pragma unroll
  for (int kc = 0; kc < 3; ++kc)
    __builtin_prefetch(WoutP + (((kc * 128) + wave * 16) * 32) * 16, 0, 1);

  for (int c = tid; c < 16 * KCTX / 8; c += 256)
    *(v8h*)&sA[c * 8] = *(const v8h*)&ctx[r0 * KCTX + c * 8];
  __syncthreads();

  // A fragments built once per wave, reused across all 16 N-tiles
  v16h a0 = load_a_frag(sA, KCTX, 0, lane);
  v16h a1 = load_a_frag(sA, KCTX, 32, lane);
  v16h a2 = load_a_frag(sA, KCTX, 64, lane);

  for (int q = 0; q < 16; ++q) {
    int nt = wave * 16 + q;
    v8f acc = {};
    acc = __builtin_amdgcn_wmma_f32_16x16x32_f16(false, a0, false, load_b_packed(WoutP, 128, 0, nt, lane), (short)0, acc, false, false);
    acc = __builtin_amdgcn_wmma_f32_16x16x32_f16(false, a1, false, load_b_packed(WoutP, 128, 1, nt, lane), (short)0, acc, false, false);
    acc = __builtin_amdgcn_wmma_f32_16x16x32_f16(false, a2, false, load_b_packed(WoutP, 128, 2, nt, lane), (short)0, acc, false, false);
    int n = nt * 16 + nl;
    float bo = b_out[n];
#pragma unroll
    for (int i = 0; i < 8; ++i) sLog[(i + mh) * V_OUT + n] = acc[i] + bo;
  }
  __syncthreads();

  // log_softmax per row (2 rows per wave), vectorized scan + NT streaming store
  for (int rr = wave * 2; rr < wave * 2 + 2; ++rr) {
    const float* rowp = &sLog[rr * V_OUT];
    float mx = -3.4e38f;
    for (int c = lane; c < V_OUT / 4; c += 32) {
      v4f x = *(const v4f*)&rowp[c * 4];
      mx = fmaxf(mx, fmaxf(fmaxf(x.x, x.y), fmaxf(x.z, x.w)));
    }
#pragma unroll
    for (int off = 1; off < 32; off <<= 1) mx = fmaxf(mx, __shfl_xor(mx, off, 32));
    float s = 0.0f;
    for (int c = lane; c < V_OUT / 4; c += 32) {
      v4f x = *(const v4f*)&rowp[c * 4];
      s += __expf(x.x - mx) + __expf(x.y - mx) + __expf(x.z - mx) + __expf(x.w - mx);
    }
#pragma unroll
    for (int off = 1; off < 32; off <<= 1) s += __shfl_xor(s, off, 32);
    float lz = mx + __logf(s);
    long long base = (r0 + rr) * (long long)V_OUT;
    for (int c = lane; c < V_OUT / 4; c += 32) {
      v4f x = *(const v4f*)&rowp[c * 4];
      v4f y = x - lz;
      __builtin_nontemporal_store(y, (v4f*)&out[base + c * 4]);
    }
  }
}

extern "C" void kernel_launch(void* const* d_in, const int* in_sizes, int n_in,
                              void* d_out, int out_size, void* d_ws, size_t ws_size,
                              hipStream_t stream) {
  const int*   inst   = (const int*)  d_in[0];
  const float* prev   = (const float*)d_in[1];
  const float* fin    = (const float*)d_in[2];
  const float* embed  = (const float*)d_in[3];
  const float* W_ih   = (const float*)d_in[4];
  const float* W_hh   = (const float*)d_in[5];
  const float* b_ih   = (const float*)d_in[6];
  const float* b_hh   = (const float*)d_in[7];
  const float* W_att1 = (const float*)d_in[8];
  const float* b_att1 = (const float*)d_in[9];
  const float* W_att2 = (const float*)d_in[10];
  const float* b_att2 = (const float*)d_in[11];
  const float* W_out  = (const float*)d_in[12];
  const float* b_out  = (const float*)d_in[13];
  float* out = (float*)d_out;

  char* ws = (char*)d_ws;
  size_t o = 0;
  float*    memB  = (float*)(ws + o);    o += (size_t)B_TOT * 50 * 4 * 4;        // 409600
  _Float16* WgP   = (_Float16*)(ws + o); o += (size_t)4 * 16 * 512 * 2;          // 65536
  float*    bg    = (float*)(ws + o);    o += 256 * 4;                            // 1024
  _Float16* Wa1P  = (_Float16*)(ws + o); o += (size_t)3 * 2 * 512 * 2;           // 6144
  _Float16* WoutP = (_Float16*)(ws + o); o += (size_t)3 * 128 * 512 * 2;         // 393216
  _Float16* ctx   = (_Float16*)(ws + o); o += (size_t)B_TOT * T_SEQ * KCTX * 2;  // 6291456

  prep_kernel<<<256, 256, 0, stream>>>(prev, fin, W_ih, W_hh, b_ih, b_hh, W_att1, W_out,
                                       memB, WgP, bg, Wa1P, WoutP, ctx);
  recur_kernel<<<B_TOT / 16, 256, 0, stream>>>(inst, embed, WgP, bg, Wa1P, b_att1,
                                               W_att2, b_att2, memB, ctx);
  logits_kernel<<<(B_TOT * T_SEQ) / 16, 256, 0, stream>>>(ctx, WoutP, b_out, out);
}
